// QuadPrior_Attention_52321291600384
// MI455X (gfx1250) — matrix operand order repaired and checked by reading, not compiled, not run on hardware
//
#include <hip/hip_runtime.h>
#include <hip/hip_bf16.h>
#include <math.h>

#define BATCH   2
#define SEQLEN  2048
#define EMBED   1024
#define NHEADS  16
#define HDIM    64
#define MROWS   (BATCH * SEQLEN)   /* 4096 */

typedef __bf16 bf16_t;
typedef __attribute__((ext_vector_type(16))) __bf16 v16bf;
typedef __attribute__((ext_vector_type(8)))  __bf16 v8bf;
typedef __attribute__((ext_vector_type(8)))  float  v8f;

#if defined(__has_builtin)
#if __has_builtin(__builtin_amdgcn_global_load_async_to_lds_b128) && \
    __has_builtin(__builtin_amdgcn_s_wait_asynccnt)
#define QP_ASYNC_LDS 1
#endif
#endif
#ifndef QP_ASYNC_LDS
#define QP_ASYNC_LDS 0
#endif

#if QP_ASYNC_LDS
typedef int qp_v4i __attribute__((ext_vector_type(4)));
typedef __attribute__((address_space(1))) qp_v4i* qp_gptr;   // global src
typedef __attribute__((address_space(3))) qp_v4i* qp_lptr;   // LDS dst
#endif

// ---------------------------------------------------------------------------
// fp32 -> bf16 conversion
// ---------------------------------------------------------------------------
__global__ void qp_f32_to_bf16(const float* __restrict__ in,
                               bf16_t* __restrict__ out, int n) {
    int i = blockIdx.x * blockDim.x + threadIdx.x;
    if (i < n) out[i] = (bf16_t)in[i];
}

// ---------------------------------------------------------------------------
// Build dense TRANSPOSED projection matrix Wt[n][k] (n = out dim, k = in dim)
// from block-tridiagonal (Wv: 8x128x128, Wr: 7x128x128, Wl: 7x128x128).
//   y[bo] = x[bo] @ Wv[bo] + x[bo-1] @ Wr[bo-1] + x[bo+1] @ Wl[bo]
// ---------------------------------------------------------------------------
__global__ void qp_build_wt(const float* __restrict__ Wv,
                            const float* __restrict__ Wr,
                            const float* __restrict__ Wl,
                            bf16_t* __restrict__ Wt) {
    int idx = blockIdx.x * blockDim.x + threadIdx.x;   // < EMBED*EMBED
    int n = idx >> 10;        // output dim t_global
    int k = idx & 1023;       // input dim  s_global
    int bo = n >> 7, t = n & 127;
    int bi = k >> 7, s = k & 127;
    float v = 0.0f;
    if (bi == bo)            v = Wv[((bo    ) * 128 + s) * 128 + t];
    else if (bi == bo - 1)   v = Wr[((bo - 1) * 128 + s) * 128 + t];
    else if (bi == bo + 1)   v = Wl[((bo    ) * 128 + s) * 128 + t];
    Wt[idx] = (bf16_t)v;
}

// ---------------------------------------------------------------------------
// WMMA GEMM: C[M,N] = A[M,K](bf16 row-major) * B (given as Bt[N,K] bf16)
//            + bias[N], f32 accumulate, f32 output.
//
// Block = 128 threads (4 waves). Block tile = 128(M) x 64(N).
// Wave w owns rows [w*32, w*32+32) and all 64 N columns:
//   2 A-fragments x 4 B-fragments -> 8 v_wmma per K-step of 32.
// The 32x64 B tile is shared by all 4 waves -> staged in LDS once per
// K-step (async global->LDS when available), read back with ds_load_b128.
// LDS row stride padded to 40 halfs (80B) for bank-conflict-free reads.
//
// Fragment layouts used here (wave32, contiguous k order in A and B):
//  A 16x32 bf16 : lane L -> row M = L%16 ; elems 0..7  = Kbase+16*(L/16)
//                                          elems 8..15 = Kbase+16*(L/16)+8
//  B 32x16 bf16 : lane L -> col N = L%16 ; elems 0..15 = Kbase+16*(L/16)
//  D 16x16 f32  : lane L, vgpr r -> row M = r + 8*(L/16), col N = L%16
// ---------------------------------------------------------------------------
#define BROW 40          /* LDS row stride in bf16 elems (32 payload + 8 pad) */

__global__ void qp_gemm_wmma(const bf16_t* __restrict__ A,
                             const bf16_t* __restrict__ Bt,
                             const float*  __restrict__ bias,
                             float* __restrict__ Cout,
                             int M, int N, int K) {
    __shared__ bf16_t Bs[64 * BROW];          // 5 KB

    const int tid  = threadIdx.x;
    const int lane = tid & 31;
    const int wave = tid >> 5;
    const int hf   = lane >> 4;               // 0 or 1 (half-wave)
    const int lr   = lane & 15;
    const int nblk = blockIdx.x * 64;         // block N base
    const int m0   = blockIdx.y * 128 + wave * 32;  // wave M base

    const bf16_t* arow0 = A + (size_t)(m0 + lr) * K;        // A frag 0 rows
    const bf16_t* arow1 = A + (size_t)(m0 + 16 + lr) * K;   // A frag 1 rows

    // staging assignment: thread tid copies 32B: row = tid/2, half = tid&1
    const int srow = tid >> 1;
    const int shh  = tid & 1;
    const bf16_t* sbase = Bt + (size_t)(nblk + srow) * K + shh * 16;
    bf16_t*       dbase = &Bs[srow * BROW + shh * 16];

    v8f acc[2][4];
    #pragma unroll
    for (int mm = 0; mm < 2; ++mm)
        #pragma unroll
        for (int nn = 0; nn < 4; ++nn)
            acc[mm][nn] = (v8f){};

    #pragma unroll 2
    for (int k0 = 0; k0 < K; k0 += 32) {
        // ---- stage 32(K) x 64(N) B tile into LDS ----
#if QP_ASYNC_LDS
        __builtin_amdgcn_global_load_async_to_lds_b128(
            (qp_gptr)(sbase + k0), (qp_lptr)dbase, 0, 0);
        __builtin_amdgcn_global_load_async_to_lds_b128(
            (qp_gptr)(sbase + k0 + 8), (qp_lptr)(dbase + 8), 0, 0);
        __builtin_amdgcn_s_wait_asynccnt(0);
#else
        {
            v8bf t0 = *(const v8bf*)(sbase + k0);
            v8bf t1 = *(const v8bf*)(sbase + k0 + 8);
            *(v8bf*)dbase       = t0;
            *(v8bf*)(dbase + 8) = t1;
        }
#endif
        __syncthreads();

        // prefetch next A chunk
        __builtin_prefetch(arow0 + k0 + 128, 0, 1);

        // ---- A fragments from global ----
        v16bf a0, a1;
        {
            v8bf lo = *(const v8bf*)(arow0 + k0 + hf * 16);
            v8bf hi = *(const v8bf*)(arow0 + k0 + hf * 16 + 8);
            #pragma unroll
            for (int i = 0; i < 8; ++i) { a0[i] = lo[i]; a0[8 + i] = hi[i]; }
        }
        {
            v8bf lo = *(const v8bf*)(arow1 + k0 + hf * 16);
            v8bf hi = *(const v8bf*)(arow1 + k0 + hf * 16 + 8);
            #pragma unroll
            for (int i = 0; i < 8; ++i) { a1[i] = lo[i]; a1[8 + i] = hi[i]; }
        }

        // ---- B fragments from LDS + 8 WMMAs ----
        #pragma unroll
        for (int nn = 0; nn < 4; ++nn) {
            v16bf b = *(const v16bf*)&Bs[(nn * 16 + lr) * BROW + hf * 16];
            acc[0][nn] = __builtin_amdgcn_wmma_f32_16x16x32_bf16(
                             false, a0, false, b, (short)0, acc[0][nn], false, false);
            acc[1][nn] = __builtin_amdgcn_wmma_f32_16x16x32_bf16(
                             false, a1, false, b, (short)0, acc[1][nn], false, false);
        }
        __syncthreads();
    }

    // ---- epilogue: bias + store f32 ----
    #pragma unroll
    for (int nn = 0; nn < 4; ++nn) {
        const int ncol = nblk + nn * 16 + lr;
        const float bv = bias[ncol];
        #pragma unroll
        for (int mm = 0; mm < 2; ++mm) {
            const int mbase = m0 + mm * 16 + hf * 8;
            #pragma unroll
            for (int r = 0; r < 8; ++r)
                Cout[(size_t)(mbase + r) * N + ncol] = acc[mm][nn][r] + bv;
        }
    }
}

// ---------------------------------------------------------------------------
// Sparse masked attention. mask row t allows columns t-off for
// off in {0,1,2,3,4, 8,16,32,...,1024} (<= 13 valid columns).
// One thread per (b,h,t). q,k,v are fp32 laid out as [B*T, EMBED] with
// head h occupying columns [h*64, h*64+64). Output written as bf16
// in the same [B*T, EMBED] layout (already "head-merged").
// ---------------------------------------------------------------------------
__global__ void qp_sparse_attn(const float* __restrict__ q,
                               const float* __restrict__ k,
                               const float* __restrict__ v,
                               bf16_t* __restrict__ out) {
    int gid = blockIdx.x * blockDim.x + threadIdx.x;   // < BATCH*NHEADS*SEQLEN
    int t  = gid & (SEQLEN - 1);
    int bh = gid >> 11;            // SEQLEN = 2048
    int h  = bh & (NHEADS - 1);
    int b  = bh >> 4;

    const size_t rowbase = (size_t)(b * SEQLEN + t) * EMBED + h * HDIM;
    const float* qrow = q + rowbase;

    float qv[HDIM];
    #pragma unroll
    for (int d = 0; d < HDIM; ++d) qv[d] = qrow[d];

    int offs[13];
    int ncol = 0;
    #pragma unroll
    for (int d0 = 0; d0 < 5; ++d0)
        if (t - d0 >= 0) offs[ncol++] = d0;
    #pragma unroll
    for (int st = 8; st < SEQLEN; st <<= 1)
        if (t - st >= 0) offs[ncol++] = st;

    float sc[13];
    float mx = -INFINITY;
    for (int i = 0; i < ncol; ++i) {
        int s = t - offs[i];
        const float* krow = k + (size_t)(b * SEQLEN + s) * EMBED + h * HDIM;
        float dot = 0.0f;
        #pragma unroll
        for (int d = 0; d < HDIM; ++d) dot += qv[d] * krow[d];
        sc[i] = dot * 0.125f;               // 1/sqrt(64)
        mx = fmaxf(mx, sc[i]);
    }

    float ov[HDIM];
    #pragma unroll
    for (int d = 0; d < HDIM; ++d) ov[d] = 0.0f;
    float sum = 0.0f;
    for (int i = 0; i < ncol; ++i) {
        int s = t - offs[i];
        float w = __expf(sc[i] - mx);
        sum += w;
        const float* vrow = v + (size_t)(b * SEQLEN + s) * EMBED + h * HDIM;
        #pragma unroll
        for (int d = 0; d < HDIM; ++d) ov[d] += w * vrow[d];
    }

    float inv = 1.0f / sum;
    bf16_t* orow = out + rowbase;
    #pragma unroll
    for (int d = 0; d < HDIM; ++d) orow[d] = (bf16_t)(ov[d] * inv);
}

// ---------------------------------------------------------------------------
// Launch
// d_in order: x, q_Wv,q_Wr,q_Wl,q_b, k_Wv,k_Wr,k_Wl,k_b,
//             v_Wv,v_Wr,v_Wl,v_b, o_Wv,o_Wr,o_Wl,o_b
// ---------------------------------------------------------------------------
extern "C" void kernel_launch(void* const* d_in, const int* in_sizes, int n_in,
                              void* d_out, int out_size, void* d_ws, size_t ws_size,
                              hipStream_t stream) {
    (void)in_sizes; (void)n_in; (void)out_size; (void)ws_size;

    const float* x = (const float*)d_in[0];
    const float* Wv[4], * Wr[4], * Wl[4], * bias[4];
    for (int p = 0; p < 4; ++p) {
        Wv[p]   = (const float*)d_in[1 + 4 * p + 0];
        Wr[p]   = (const float*)d_in[1 + 4 * p + 1];
        Wl[p]   = (const float*)d_in[1 + 4 * p + 2];
        bias[p] = (const float*)d_in[1 + 4 * p + 3];
    }

    // ---- workspace layout ----
    char*  ws   = (char*)d_ws;
    size_t off  = 0;
    bf16_t* xbf = (bf16_t*)(ws + off); off += (size_t)MROWS * EMBED * sizeof(bf16_t);
    bf16_t* wt  = (bf16_t*)(ws + off); off += (size_t)4 * EMBED * EMBED * sizeof(bf16_t);
    float*  qf  = (float*) (ws + off); off += (size_t)MROWS * EMBED * sizeof(float);
    float*  kf  = (float*) (ws + off); off += (size_t)MROWS * EMBED * sizeof(float);
    float*  vf  = (float*) (ws + off); off += (size_t)MROWS * EMBED * sizeof(float);
    bf16_t* abf = (bf16_t*)(ws + off); off += (size_t)MROWS * EMBED * sizeof(bf16_t);

    // 1) x -> bf16
    {
        int n = MROWS * EMBED;
        qp_f32_to_bf16<<<(n + 255) / 256, 256, 0, stream>>>(x, xbf, n);
    }

    // 2) pack dense transposed weight matrices (bf16), order q,k,v,o
    for (int p = 0; p < 4; ++p) {
        int n = EMBED * EMBED;
        qp_build_wt<<<(n + 255) / 256, 256, 0, stream>>>(
            Wv[p], Wr[p], Wl[p], wt + (size_t)p * EMBED * EMBED);
    }

    // 3) q,k,v projections (WMMA bf16 -> f32)
    dim3 ggrid(EMBED / 64, MROWS / 128);
    qp_gemm_wmma<<<ggrid, 128, 0, stream>>>(xbf, wt + 0 * (size_t)EMBED * EMBED,
                                            bias[0], qf, MROWS, EMBED, EMBED);
    qp_gemm_wmma<<<ggrid, 128, 0, stream>>>(xbf, wt + 1 * (size_t)EMBED * EMBED,
                                            bias[1], kf, MROWS, EMBED, EMBED);
    qp_gemm_wmma<<<ggrid, 128, 0, stream>>>(xbf, wt + 2 * (size_t)EMBED * EMBED,
                                            bias[2], vf, MROWS, EMBED, EMBED);

    // 4) sparse attention (<= 13 columns/row), output bf16 head-merged
    {
        int n = BATCH * NHEADS * SEQLEN;    // 65536
        qp_sparse_attn<<<(n + 255) / 256, 256, 0, stream>>>(qf, kf, vf, abf);
    }

    // 5) output projection (WMMA bf16 -> f32) into d_out
    qp_gemm_wmma<<<ggrid, 128, 0, stream>>>(abf, wt + 3 * (size_t)EMBED * EMBED,
                                            bias[3], (float*)d_out,
                                            MROWS, EMBED, EMBED);
}
